// MultiHeadAttention_30923764532017
// MI455X (gfx1250) — matrix operand order repaired and checked by reading, not compiled, not run on hardware
//
#include <hip/hip_runtime.h>
#include <hip/hip_bf16.h>
#include <math.h>

// ---------------------------------------------------------------------------
// Problem constants (match reference)
// ---------------------------------------------------------------------------
#define D_MODEL 1024
#define N_HEAD  16
#define DK      64          // head dim
#define BATCH   4
#define SEQ     2048
#define NTOK    (BATCH * SEQ)   // 8192 tokens

typedef __attribute__((ext_vector_type(16))) _Float16 v16h;
typedef __attribute__((ext_vector_type(8)))  _Float16 v8h;
typedef __attribute__((ext_vector_type(8)))  float    v8f;

union FragAB { v16h v; v8h h[2]; };

__device__ __forceinline__ v8f wmma_f16(v16h a, v16h b, v8f c) {
  // D = A(16x32 f16) * B(32x16 f16) + C(16x16 f32)
  return __builtin_amdgcn_wmma_f32_16x16x32_f16(
      /*neg_a=*/false, a, /*neg_b=*/false, b,
      /*c_mod=*/(short)0, c, /*reuse_a=*/false, /*reuse_b=*/false);
}

// ---------------------------------------------------------------------------
// fp32 -> fp16 conversion (grid-stride)
// ---------------------------------------------------------------------------
__global__ __launch_bounds__(256) void cvt_f32_to_f16(
    const float* __restrict__ src, _Float16* __restrict__ dst, int n) {
  int i = blockIdx.x * blockDim.x + threadIdx.x;
  int stride = gridDim.x * blockDim.x;
  for (; i < n; i += stride) dst[i] = (_Float16)src[i];
}

// ---------------------------------------------------------------------------
// QKV GEMM:  QKV[n, m] = sum_c x[n,c] * w_attn[m,c] + b_attn[m]
//   n in [0, 8192), m in [0, 3072).
// One wave per 32x32 output tile.  Scatter results into
//   Q  [B,H,T,dk]  (pre-scaled by 1/sqrt(dk))
//   K  [B,H,T,dk]
//   Vt [B,H,dk,T]  (transposed for contiguous P*V B-fragments)
// ---------------------------------------------------------------------------
__global__ __launch_bounds__(256) void qkv_gemm(
    const _Float16* __restrict__ xh,    // [NTOK][D_MODEL]
    const _Float16* __restrict__ wah,   // [3*D_MODEL][D_MODEL]
    const float*    __restrict__ b_attn,
    _Float16* __restrict__ qbuf,
    _Float16* __restrict__ kbuf,
    _Float16* __restrict__ vtbuf) {
  const int lane = threadIdx.x & 31;
  const int hl   = lane >> 4;     // lane half (0/1)
  const int lr   = lane & 15;     // lane within half
  const int wave = blockIdx.x * (blockDim.x >> 5) + (threadIdx.x >> 5);
  const int ntn  = (3 * D_MODEL) / 32;      // 96 N-tiles
  const int tile_m = wave / ntn;            // 0..255
  const int tile_n = wave % ntn;
  if (tile_m >= NTOK / 32) return;

  v8f acc[2][2] = {};
  for (int k0 = 0; k0 < D_MODEL; k0 += 32) {
    FragAB a[2], b[2];
#pragma unroll
    for (int mi = 0; mi < 2; ++mi) {
      const _Float16* p = xh + (size_t)(tile_m * 32 + mi * 16 + lr) * D_MODEL + k0;
      // A layout (16-bit, 16x32): K(e) = (e/8)*16 + hl*8 + e%8
      a[mi].h[0] = *(const v8h*)(p + hl * 8);
      a[mi].h[1] = *(const v8h*)(p + 16 + hl * 8);
      __builtin_prefetch(p + 32, 0, 0);
    }
#pragma unroll
    for (int ni = 0; ni < 2; ++ni) {
      const _Float16* p = wah + (size_t)(tile_n * 32 + ni * 16 + lr) * D_MODEL + k0;
      // B layout (16-bit, 32x16): K(e) = (e/8)*16 + hl*8 + e%8, N = lr
      b[ni].h[0] = *(const v8h*)(p + hl * 8);
      b[ni].h[1] = *(const v8h*)(p + 16 + hl * 8);
      __builtin_prefetch(p + 32, 0, 0);
    }
#pragma unroll
    for (int mi = 0; mi < 2; ++mi)
#pragma unroll
      for (int ni = 0; ni < 2; ++ni)
        acc[mi][ni] = wmma_f16(a[mi].v, b[ni].v, acc[mi][ni]);
  }

  const float qscale = 0.125f;   // 1/sqrt(64), folded into Q
#pragma unroll
  for (int ni = 0; ni < 2; ++ni) {
    const int m = tile_n * 32 + ni * 16 + lr;   // channel in [0, 3C)
    const float bias = b_attn[m];
#pragma unroll
    for (int mi = 0; mi < 2; ++mi) {
#pragma unroll
      for (int v = 0; v < 8; ++v) {
        // D layout: row M = v + 8*hl, col N = lr
        const int n  = tile_m * 32 + mi * 16 + v + hl * 8;  // token
        const int bb = n / SEQ, t = n % SEQ;
        const float val = acc[mi][ni][v] + bias;
        if (m < D_MODEL) {
          const int h = m / DK, d = m % DK;
          qbuf[((size_t)(bb * N_HEAD + h) * SEQ + t) * DK + d] =
              (_Float16)(val * qscale);
        } else if (m < 2 * D_MODEL) {
          const int c = m - D_MODEL, h = c / DK, d = c % DK;
          kbuf[((size_t)(bb * N_HEAD + h) * SEQ + t) * DK + d] = (_Float16)val;
        } else {
          const int c = m - 2 * D_MODEL, h = c / DK, d = c % DK;
          vtbuf[((size_t)(bb * N_HEAD + h) * DK + d) * SEQ + t] = (_Float16)val;
        }
      }
    }
  }
}

// ---------------------------------------------------------------------------
// Attention: one wave per (b, h, 16-query tile).  Flash-style online softmax
// streaming 32 keys per iteration.  S/P transpose (D-layout -> A-layout) goes
// through a per-wave LDS tile, ordered with s_wait_dscnt.
// ---------------------------------------------------------------------------
__global__ __launch_bounds__(256) void attn_kernel(
    const _Float16* __restrict__ qbuf,   // [B,H,T,DK] (pre-scaled)
    const _Float16* __restrict__ kbuf,   // [B,H,T,DK]
    const _Float16* __restrict__ vtbuf,  // [B,H,DK,T]
    _Float16* __restrict__ ybuf) {       // [B,T,C]
  __shared__ _Float16 lds_p[8][16 * 32];   // per-wave 16x32 P tile

  const int lane  = threadIdx.x & 31;
  const int hl    = lane >> 4;
  const int lr    = lane & 15;
  const int wslot = threadIdx.x >> 5;
  const int wave  = blockIdx.x * (blockDim.x >> 5) + wslot;

  const int bh = wave >> 7;            // (b*16 + h), 0..63
  const int q0 = (wave & 127) * 16;    // query tile start
  const int bb = bh >> 4;
  const int hh = bh & 15;

  const _Float16* Qb = qbuf  + (size_t)bh * SEQ * DK;
  const _Float16* Kb = kbuf  + (size_t)bh * SEQ * DK;
  const _Float16* Vb = vtbuf + (size_t)bh * DK * SEQ;

  // Resident Q fragments: A(16x32) per 32-wide chunk of dk=64
  FragAB aq[2];
#pragma unroll
  for (int cc = 0; cc < 2; ++cc) {
    const _Float16* p = Qb + (size_t)(q0 + lr) * DK + cc * 32;
    aq[cc].h[0] = *(const v8h*)(p + hl * 8);
    aq[cc].h[1] = *(const v8h*)(p + 16 + hl * 8);
  }

  v8f o[4] = {};
  float rmax[8], rsum[8];
#pragma unroll
  for (int v = 0; v < 8; ++v) { rmax[v] = -1e30f; rsum[v] = 0.f; }

  _Float16* myp = lds_p[wslot];

  for (int kt = 0; kt < SEQ; kt += 32) {
    // ---- scores: S0 = Q @ K[kt..kt+15]^T, S1 = Q @ K[kt+16..kt+31]^T ----
    v8f s0 = {}; v8f s1 = {};
#pragma unroll
    for (int cc = 0; cc < 2; ++cc) {
      FragAB bk;
      const _Float16* p = Kb + (size_t)(kt + lr) * DK + cc * 32;
      bk.h[0] = *(const v8h*)(p + hl * 8);
      bk.h[1] = *(const v8h*)(p + 16 + hl * 8);
      s0 = wmma_f16(aq[cc].v, bk.v, s0);
      p = Kb + (size_t)(kt + 16 + lr) * DK + cc * 32;
      bk.h[0] = *(const v8h*)(p + hl * 8);
      bk.h[1] = *(const v8h*)(p + 16 + hl * 8);
      s1 = wmma_f16(aq[cc].v, bk.v, s1);
    }

    // ---- online softmax (row = v + 8*hl, cols across the 16-lane half) ----
    float pl0[8], pl1[8];
#pragma unroll
    for (int v = 0; v < 8; ++v) {
      float m = fmaxf(s0[v], s1[v]);
#pragma unroll
      for (int x = 1; x < 16; x <<= 1) m = fmaxf(m, __shfl_xor(m, x, 32));
      const float mnew = fmaxf(rmax[v], m);
      const float corr = __expf(rmax[v] - mnew);
      rmax[v] = mnew;
      const float p0 = __expf(s0[v] - mnew);
      const float p1 = __expf(s1[v] - mnew);
      float psum = p0 + p1;
#pragma unroll
      for (int x = 1; x < 16; x <<= 1) psum += __shfl_xor(psum, x, 32);
      rsum[v] = rsum[v] * corr + psum;
      pl0[v] = p0; pl1[v] = p1;
#pragma unroll
      for (int ni = 0; ni < 4; ++ni) o[ni][v] *= corr;
    }

    // ---- transpose P (D-layout) -> A-layout via per-wave LDS tile ----
#pragma unroll
    for (int v = 0; v < 8; ++v) {
      myp[(v + hl * 8) * 32 + lr]      = (_Float16)pl0[v];
      myp[(v + hl * 8) * 32 + 16 + lr] = (_Float16)pl1[v];
    }
    asm volatile("s_wait_dscnt 0" ::: "memory");  // wave-local LDS RAW order
    FragAB pf;
    {
      const _Float16* p = myp + lr * 32;
      pf.h[0] = *(const v8h*)(p + hl * 8);
      pf.h[1] = *(const v8h*)(p + 16 + hl * 8);
    }

    // ---- O += P(16x32) @ V(32x64); V^T layout -> contiguous B frags ----
#pragma unroll
    for (int ni = 0; ni < 4; ++ni) {
      FragAB bv;
      const _Float16* p = Vb + (size_t)(ni * 16 + lr) * SEQ + kt;
      bv.h[0] = *(const v8h*)(p + hl * 8);
      bv.h[1] = *(const v8h*)(p + 16 + hl * 8);
      o[ni] = wmma_f16(pf.v, bv.v, o[ni]);
    }
  }

  // ---- normalize and write y_att [B,T,C] ----
#pragma unroll
  for (int v = 0; v < 8; ++v) rsum[v] = 1.0f / rsum[v];
#pragma unroll
  for (int ni = 0; ni < 4; ++ni) {
#pragma unroll
    for (int v = 0; v < 8; ++v) {
      const int t = q0 + v + hl * 8;
      const int c = hh * DK + ni * 16 + lr;
      ybuf[((size_t)(bb * SEQ + t)) * D_MODEL + c] =
          (_Float16)(o[ni][v] * rsum[v]);
    }
  }
}

// ---------------------------------------------------------------------------
// Output projection: out[n, m] = sum_c y[n,c] * w_proj[m,c] + b_proj[m] (f32)
// ---------------------------------------------------------------------------
__global__ __launch_bounds__(256) void proj_gemm(
    const _Float16* __restrict__ yh,    // [NTOK][D_MODEL]
    const _Float16* __restrict__ wph,   // [D_MODEL][D_MODEL]
    const float*    __restrict__ b_proj,
    float* __restrict__ out) {          // [NTOK][D_MODEL]
  const int lane = threadIdx.x & 31;
  const int hl   = lane >> 4;
  const int lr   = lane & 15;
  const int wave = blockIdx.x * (blockDim.x >> 5) + (threadIdx.x >> 5);
  const int ntn  = D_MODEL / 32;        // 32 N-tiles
  const int tile_m = wave / ntn;
  const int tile_n = wave % ntn;
  if (tile_m >= NTOK / 32) return;

  v8f acc[2][2] = {};
  for (int k0 = 0; k0 < D_MODEL; k0 += 32) {
    FragAB a[2], b[2];
#pragma unroll
    for (int mi = 0; mi < 2; ++mi) {
      const _Float16* p = yh + (size_t)(tile_m * 32 + mi * 16 + lr) * D_MODEL + k0;
      a[mi].h[0] = *(const v8h*)(p + hl * 8);
      a[mi].h[1] = *(const v8h*)(p + 16 + hl * 8);
      __builtin_prefetch(p + 32, 0, 0);
    }
#pragma unroll
    for (int ni = 0; ni < 2; ++ni) {
      const _Float16* p = wph + (size_t)(tile_n * 32 + ni * 16 + lr) * D_MODEL + k0;
      b[ni].h[0] = *(const v8h*)(p + hl * 8);
      b[ni].h[1] = *(const v8h*)(p + 16 + hl * 8);
      __builtin_prefetch(p + 32, 0, 0);
    }
#pragma unroll
    for (int mi = 0; mi < 2; ++mi)
#pragma unroll
      for (int ni = 0; ni < 2; ++ni)
        acc[mi][ni] = wmma_f16(a[mi].v, b[ni].v, acc[mi][ni]);
  }

#pragma unroll
  for (int ni = 0; ni < 2; ++ni) {
    const int m = tile_n * 32 + ni * 16 + lr;
    const float bias = b_proj[m];
#pragma unroll
    for (int mi = 0; mi < 2; ++mi) {
#pragma unroll
      for (int v = 0; v < 8; ++v) {
        const int n = tile_m * 32 + mi * 16 + v + hl * 8;
        out[(size_t)n * D_MODEL + m] = acc[mi][ni][v] + bias;
      }
    }
  }
}

// ---------------------------------------------------------------------------
// Launch
// ---------------------------------------------------------------------------
extern "C" void kernel_launch(void* const* d_in, const int* in_sizes, int n_in,
                              void* d_out, int out_size, void* d_ws, size_t ws_size,
                              hipStream_t stream) {
  const float* x      = (const float*)d_in[0];
  const float* w_attn = (const float*)d_in[1];
  const float* b_attn = (const float*)d_in[2];
  const float* w_proj = (const float*)d_in[3];
  const float* b_proj = (const float*)d_in[4];

  char* ws = (char*)d_ws;
  _Float16* xh  = (_Float16*)ws;  ws += (size_t)NTOK * D_MODEL * 2;
  _Float16* wah = (_Float16*)ws;  ws += (size_t)3 * D_MODEL * D_MODEL * 2;
  _Float16* wph = (_Float16*)ws;  ws += (size_t)D_MODEL * D_MODEL * 2;
  _Float16* qb  = (_Float16*)ws;  ws += (size_t)NTOK * D_MODEL * 2;
  _Float16* kb  = (_Float16*)ws;  ws += (size_t)NTOK * D_MODEL * 2;
  _Float16* vtb = (_Float16*)ws;  ws += (size_t)NTOK * D_MODEL * 2;
  _Float16* yb  = (_Float16*)ws;

  // 1) fp32 -> fp16 conversions
  cvt_f32_to_f16<<<2048, 256, 0, stream>>>(x,      xh,  NTOK * D_MODEL);
  cvt_f32_to_f16<<<1024, 256, 0, stream>>>(w_attn, wah, 3 * D_MODEL * D_MODEL);
  cvt_f32_to_f16<<<512,  256, 0, stream>>>(w_proj, wph, D_MODEL * D_MODEL);

  // 2) QKV GEMM: 256 x 96 tiles of 32x32, 8 waves/block
  qkv_gemm<<<(256 * 96) / 8, 256, 0, stream>>>(xh, wah, b_attn, qb, kb, vtb);

  // 3) Attention: 64 (b,h) x 128 q-tiles, 8 waves/block
  attn_kernel<<<(64 * 128) / 8, 256, 0, stream>>>(qb, kb, vtb, yb);

  // 4) Projection: 256 x 32 tiles of 32x32, 8 waves/block
  proj_gemm<<<(256 * 32) / 8, 256, 0, stream>>>(yb, wph, b_proj, (float*)d_out);
}
